// GraphScoreNet_34479997453189
// MI455X (gfx1250) — matrix operand order repaired and checked by reading, not compiled, not run on hardware
//
#include <hip/hip_runtime.h>
#include <hip/hip_bf16.h>
#include <math.h>

typedef __bf16 bf16_t;
typedef __attribute__((ext_vector_type(16))) __bf16 v16bf;
typedef __attribute__((ext_vector_type(8)))  __bf16 v8bf;
typedef __attribute__((ext_vector_type(8)))  float  v8f;
typedef unsigned int u32;
typedef __attribute__((ext_vector_type(4))) u32 u32x4;
typedef __attribute__((ext_vector_type(8))) int i32x8;
typedef __attribute__((ext_vector_type(4))) int i32x4;

#define B_   8
#define N_   4096
#define DF   7
#define KNN  20
#define DEMB 8
#define LAT  128
#define KP1  288           // 264 padded to 9*32
#define TILE 128
#define EPB  (N_*KNN)      // 81920 edges per batch
#define MLPE (KP1*128 + 3*128*128)  // bf16 elems per converted MLP (86016)

// LDS layout: [panel0 | panel1 | panel2 | panel3 | X tile | ridx]
#define P0_ELEMS   (128*KP1)                 // 36864
#define PL_ELEMS   (128*LAT)                 // 16384
#define WALL_BYTES ((size_t)MLPE * 2)        // 172032: all 4 weight panels
#define X_BYTES    ((size_t)TILE * KP1 * 2)  // 73728: activation tile

__device__ __forceinline__ float gelu_f(float x) {
  float u = 0.7978845608028654f * (x + 0.044715f * x * x * x);
  return 0.5f * x * (1.0f + tanhf(u));
}

// ---- TDM: issue DMA of a contiguous bf16 panel (nelem elements) global -> LDS ----
__device__ __forceinline__ void tdm_issue(const bf16_t* gsrc, u32 lds_off, u32 nelem) {
#if __has_builtin(__builtin_amdgcn_tensor_load_to_lds)
  unsigned long long ga = (unsigned long long)(uintptr_t)gsrc;
  u32x4 g0;
  g0[0] = 1u;                                             // count=1, user mode
  g0[1] = lds_off;                                        // lds_addr (bytes)
  g0[2] = (u32)ga;                                        // global_addr[31:0]
  g0[3] = (u32)((ga >> 32) & 0x01FFFFFFu) | (2u << 30);   // addr[56:32] | type=2
  i32x8 g1;
  g1[0] = (int)(1u << 16);                                // data_size=2 bytes
  g1[1] = (int)((nelem & 0xFFFFu) << 16);                 // tensor_dim0[15:0]
  g1[2] = (int)((nelem >> 16) | (1u << 16));              // tensor_dim0[31:16] | tensor_dim1=1
  g1[3] = (int)((nelem & 0xFFFFu) << 16);                 // tile_dim0 (single row tile)
  g1[4] = 1;                                              // tile_dim1=1, tile_dim2=0
  g1[5] = (int)nelem;                                     // tensor_dim0_stride[31:0]
  g1[6] = 0;
  g1[7] = 0;
  i32x4 z4 = {0, 0, 0, 0};
#if defined(__clang_major__) && (__clang_major__ >= 23)
  i32x8 z8 = {0, 0, 0, 0, 0, 0, 0, 0};
  __builtin_amdgcn_tensor_load_to_lds(g0, g1, z4, z4, z8, 0);
#else
  __builtin_amdgcn_tensor_load_to_lds(g0, g1, z4, z4, 0);
#endif
#else
  (void)gsrc; (void)lds_off; (void)nelem;
#endif
}

// ---------------- cond MLP: temb(t) ++ conditioning -> [B,8] ----------------
__global__ void k_cond(const float* t, const float* c_in,
                       const float* w0, const float* b0,
                       const float* w1, const float* b1,
                       const float* w2, const float* b2,
                       float* cond_out) {
  int b = threadIdx.x;
  if (b >= B_) return;
  float x[34];
  float targ = t[b] * 1000.0f;
  const float lf = -logf(10000.0f) / 15.0f;
  #pragma unroll
  for (int i = 0; i < 16; ++i) {
    float a = targ * expf(lf * (float)i);
    x[i]      = sinf(a);
    x[16 + i] = cosf(a);
  }
  x[32] = c_in[b*2 + 0];
  x[33] = c_in[b*2 + 1];
  float y[32];
  for (int o = 0; o < 32; ++o) {
    float s = b0[o];
    #pragma unroll
    for (int i = 0; i < 34; ++i) s += x[i] * w0[i*32 + o];
    y[o] = gelu_f(s);
  }
  float y2[32];
  for (int o = 0; o < 32; ++o) {
    float s = b1[o];
    #pragma unroll
    for (int i = 0; i < 32; ++i) s += y[i] * w1[i*32 + o];
    y2[o] = gelu_f(s);
  }
  for (int o = 0; o < 8; ++o) {
    float s = b2[o];
    #pragma unroll
    for (int i = 0; i < 32; ++i) s += y2[i] * w2[i*32 + o];
    cond_out[b*8 + o] = s;
  }
}

// ---------------- kNN: top-20 nearest (ascending), self at slot 0 ----------------
__global__ void k_knn(const float* z, int* senders, int* nbr) {
  __shared__ float sx[256], sy[256], sz[256];
  const int b = blockIdx.y;
  const int i = blockIdx.x * 256 + threadIdx.x;
  const float* zb = z + (size_t)b * N_ * DF;
  const float px = zb[i*DF+0], py = zb[i*DF+1], pz = zb[i*DF+2];
  float dk[KNN]; int ik[KNN];
  #pragma unroll
  for (int q = 0; q < KNN; ++q) { dk[q] = 3.0e38f; ik[q] = 0; }
  for (int tb = 0; tb < N_; tb += 256) {
    __syncthreads();
    {
      int j = tb + threadIdx.x;
      sx[threadIdx.x] = zb[j*DF+0];
      sy[threadIdx.x] = zb[j*DF+1];
      sz[threadIdx.x] = zb[j*DF+2];
    }
    __syncthreads();
    for (int jj = 0; jj < 256; ++jj) {
      float dx = px - sx[jj], dy = py - sy[jj], dz = pz - sz[jj];
      float d2 = dx*dx + dy*dy + dz*dz;
      int j = tb + jj;
      if (d2 < dk[KNN-1]) {
        #pragma unroll
        for (int q = KNN-1; q >= 1; --q) {
          bool ins = (d2 >= dk[q-1]);
          float nd = ins ? d2 : dk[q-1];
          int   ni = ins ? j  : ik[q-1];
          bool  wr = d2 < dk[q];
          dk[q] = wr ? nd : dk[q];
          ik[q] = wr ? ni : ik[q];
        }
        if (d2 < dk[0]) { dk[0] = d2; ik[0] = j; }
      }
    }
  }
  senders[(size_t)b*N_ + i] = ik[0];
  #pragma unroll
  for (int q = 0; q < KNN; ++q) nbr[((size_t)b*N_ + i)*KNN + q] = ik[q];
}

// ---------------- encoder: h = z @ Wenc + b ----------------
__global__ void k_encode(const float* z, const float* ew, const float* eb,
                         float* h, bf16_t* hb) {
  size_t gid = (size_t)blockIdx.x * blockDim.x + threadIdx.x;
  if (gid >= (size_t)B_ * N_ * LAT) return;
  int c = (int)(gid & (LAT - 1));
  size_t nn = gid >> 7;
  const float* zr = z + nn * DF;
  float s = eb[c];
  #pragma unroll
  for (int k = 0; k < DF; ++k) s += zr[k] * ew[k*LAT + c];
  h[gid]  = s;
  hb[gid] = (bf16_t)s;
}

// ---------------- weight convert: f32 [K,128] -> bf16 transposed [128,Kpad] ----------------
__global__ void k_wconv(const float* W, bf16_t* Wt, int K, int Kpad) {
  int gid = blockIdx.x * blockDim.x + threadIdx.x;
  if (gid >= 128 * Kpad) return;
  int c = gid / Kpad, kp = gid - c * Kpad;
  float v = (kp < K) ? W[kp*128 + c] : 0.0f;
  Wt[gid] = (bf16_t)v;
}

// ---- one dense layer on the LDS tile: X[128xKl] @ Wp[128xKl]^T (+bias, gelu/epilogue) ----
// last==false: bias+gelu, write back to X (cols 0..127).
// last==true : mode 0 -> scatter-add into aggW[ridx[row]]; mode 1 -> skip-add into hB/hbW.
__device__ __forceinline__ void mlp_layer(
    bf16_t* X, const bf16_t* Wp, const float* bl, const int Kl,
    const int wbase, const int lane, const bool last, const int mode,
    const int* ridx, float* aggW, float* hB, bf16_t* hbW)
{
  const int mrow   = lane & 15;
  const int khalf  = (lane >> 4) << 3;          // A: 0 or 8
  const int ccol   = lane & 15;                 // B/C column within 16-tile
  const int rshift = (lane < 16) ? 0 : 8;       // C row shift

  v8f acc[8];
  #pragma unroll
  for (int nt = 0; nt < 8; ++nt)
    #pragma unroll
    for (int j = 0; j < 8; ++j) acc[nt][j] = 0.0f;

  for (int kt = 0; kt < Kl; kt += 32) {
    const bf16_t* xrow = X + (size_t)(wbase + mrow) * KP1 + kt + khalf;
    v8bf alo = *(const v8bf*)(xrow);
    v8bf ahi = *(const v8bf*)(xrow + 16);
    v16bf a;
    #pragma unroll
    for (int q = 0; q < 8; ++q) { a[q] = alo[q]; a[8+q] = ahi[q]; }
    #pragma unroll
    for (int nt = 0; nt < 8; ++nt) {
      const bf16_t* wcol = Wp + (size_t)(nt*16 + ccol) * Kl + kt + ((lane < 16) ? 0 : 8);
      v8bf blo = *(const v8bf*)(wcol);
      v8bf bhi = *(const v8bf*)(wcol + 16);
      v16bf bm;
      #pragma unroll
      for (int q = 0; q < 8; ++q) { bm[q] = blo[q]; bm[8+q] = bhi[q]; }
      acc[nt] = __builtin_amdgcn_wmma_f32_16x16x32_bf16(
          false, a, false, bm, (short)0, acc[nt], false, false);
    }
  }

  if (!last) {
    #pragma unroll
    for (int nt = 0; nt < 8; ++nt) {
      int col = nt*16 + ccol;
      float bcol = bl[col];
      #pragma unroll
      for (int j = 0; j < 8; ++j) {
        float v = gelu_f(acc[nt][j] + bcol);
        int row = wbase + j + rshift;
        X[(size_t)row * KP1 + col] = (bf16_t)v;
      }
    }
  } else if (mode == 0) {
    #pragma unroll
    for (int nt = 0; nt < 8; ++nt) {
      int col = nt*16 + ccol;
      float bcol = bl[col];
      #pragma unroll
      for (int j = 0; j < 8; ++j) {
        float v = acc[nt][j] + bcol;
        int row = wbase + j + rshift;
        int recv = ridx[row];
        atomicAdd(&aggW[(size_t)recv * LAT + col], v);
      }
    }
  } else {
    #pragma unroll
    for (int nt = 0; nt < 8; ++nt) {
      int col = nt*16 + ccol;
      float bcol = bl[col];
      #pragma unroll
      for (int j = 0; j < 8; ++j) {
        int row = wbase + j + rshift;
        int n = ridx[row];
        float v = acc[nt][j] + bcol + hB[(size_t)n * LAT + col];
        hB[(size_t)n * LAT + col]  = v;
        hbW[(size_t)n * LAT + col] = (bf16_t)v;
      }
    }
  }
}

// ---------------- fused 4-layer MLP (edge or node), bf16 WMMA + pipelined TDM ----------------
__global__ void __launch_bounds__(256)
k_mlp(int mode,
      const int* senders, const int* nbr,
      const bf16_t* hb, const float* agg_in, const float* cond,
      const bf16_t* Wt,
      const float* bias0, const float* bias1, const float* bias2, const float* bias3,
      float* agg_out, float* h_out, bf16_t* hb_out)
{
  extern __shared__ char smem[];
  bf16_t* P0   = (bf16_t*)smem;                           // layer-0 panel [128][KP1]
  bf16_t* P1   = (bf16_t*)(smem + (size_t)P0_ELEMS*2);
  bf16_t* P2   = P1 + PL_ELEMS;
  bf16_t* P3   = P2 + PL_ELEMS;
  bf16_t* X    = (bf16_t*)(smem + WALL_BYTES);            // [TILE][KP1]
  int*    ridx = (int*)(smem + WALL_BYTES + X_BYTES);
  const int b    = blockIdx.y;
  const int tid  = threadIdx.x;
  const int lane = tid & 31;
  const int wv   = tid >> 5;
  const int wbase = wv * 16;

  const bf16_t* hbB  = hb     + (size_t)b * N_ * LAT;
  const float*  aggB = agg_in + (size_t)b * N_ * LAT;
  const float*  cdB  = cond   + b * DEMB;
  float*  aggW = agg_out + (size_t)b * N_ * LAT;
  float*  hB   = h_out   + (size_t)b * N_ * LAT;
  bf16_t* hbW  = hb_out  + (size_t)b * N_ * LAT;

  // ---- issue all 4 weight-panel DMAs up front (wave 0; completes in order) ----
#if __has_builtin(__builtin_amdgcn_tensor_load_to_lds)
  const u32 lds0 = (u32)__builtin_amdgcn_groupstaticsize();
  if (tid < 32) {
    tdm_issue(Wt,                          lds0,                                   P0_ELEMS);
    tdm_issue(Wt + P0_ELEMS,               lds0 + (u32)P0_ELEMS*2,                 PL_ELEMS);
    tdm_issue(Wt + P0_ELEMS + PL_ELEMS,    lds0 + (u32)(P0_ELEMS + PL_ELEMS)*2,    PL_ELEMS);
    tdm_issue(Wt + P0_ELEMS + 2*PL_ELEMS,  lds0 + (u32)(P0_ELEMS + 2*PL_ELEMS)*2,  PL_ELEMS);
  }
#endif

  // ---- gather tile into LDS (bf16); overlaps with TDM traffic ----
  {
    int r    = tid >> 1;
    int half = tid & 1;
    bf16_t* xr = X + (size_t)r * KP1;
    if (mode == 0) {
      int eg   = blockIdx.x * TILE + r;
      int i    = eg / KNN;
      int recv = nbr[(size_t)b * N_ * KNN + eg];
      int send = senders[(size_t)b * N_ + i];
      if (half == 0) {
        const uint4* src = (const uint4*)(hbB + (size_t)send * LAT);
        uint4* dst = (uint4*)xr;
        #pragma unroll
        for (int q = 0; q < 16; ++q) dst[q] = src[q];
        ridx[r] = recv;
      } else {
        const uint4* src = (const uint4*)(hbB + (size_t)recv * LAT);
        uint4* dst = (uint4*)(xr + LAT);
        #pragma unroll
        for (int q = 0; q < 16; ++q) dst[q] = src[q];
        #pragma unroll
        for (int q = 0; q < DEMB; ++q) xr[2*LAT + q] = (bf16_t)cdB[q];
        #pragma unroll
        for (int q = 2*LAT + DEMB; q < KP1; ++q) xr[q] = (bf16_t)0.0f;
      }
    } else {
      int n = blockIdx.x * TILE + r;
      if (half == 0) {
        const uint4* src = (const uint4*)(hbB + (size_t)n * LAT);
        uint4* dst = (uint4*)xr;
        #pragma unroll
        for (int q = 0; q < 16; ++q) dst[q] = src[q];
        ridx[r] = n;
      } else {
        const float* src = aggB + (size_t)n * LAT;
        #pragma unroll
        for (int q = 0; q < LAT; ++q) xr[LAT + q] = (bf16_t)src[q];
        #pragma unroll
        for (int q = 0; q < DEMB; ++q) xr[2*LAT + q] = (bf16_t)cdB[q];
        #pragma unroll
        for (int q = 2*LAT + DEMB; q < KP1; ++q) xr[q] = (bf16_t)0.0f;
      }
    }
  }
#if !__has_builtin(__builtin_amdgcn_tensor_load_to_lds)
  {
    const uint4* src = (const uint4*)Wt;
    uint4* dst = (uint4*)P0;
    for (int q = tid; q < (int)(WALL_BYTES / 16); q += 256) dst[q] = src[q];
  }
#endif
  __syncthreads();

  // ---- layer 0 (K = 288) ----
#if __has_builtin(__builtin_amdgcn_tensor_load_to_lds)
  if (tid < 32) __builtin_amdgcn_s_wait_tensorcnt(3);
  __syncthreads();
#endif
  mlp_layer(X, P0, bias0, KP1, wbase, lane, false, mode, ridx, aggW, hB, hbW);
  __syncthreads();

  // ---- layer 1 ----
#if __has_builtin(__builtin_amdgcn_tensor_load_to_lds)
  if (tid < 32) __builtin_amdgcn_s_wait_tensorcnt(2);
  __syncthreads();
#endif
  mlp_layer(X, P1, bias1, LAT, wbase, lane, false, mode, ridx, aggW, hB, hbW);
  __syncthreads();

  // ---- layer 2 ----
#if __has_builtin(__builtin_amdgcn_tensor_load_to_lds)
  if (tid < 32) __builtin_amdgcn_s_wait_tensorcnt(1);
  __syncthreads();
#endif
  mlp_layer(X, P2, bias2, LAT, wbase, lane, false, mode, ridx, aggW, hB, hbW);
  __syncthreads();

  // ---- layer 3 (epilogue: scatter-add or skip-add) ----
#if __has_builtin(__builtin_amdgcn_tensor_load_to_lds)
  if (tid < 32) __builtin_amdgcn_s_wait_tensorcnt(0);
  __syncthreads();
#endif
  mlp_layer(X, P3, bias3, LAT, wbase, lane, true, mode, ridx, aggW, hB, hbW);
}

// ---------------- decoder: out = z + h @ Wdec + b ----------------
__global__ void k_decode(const float* z, const float* h,
                         const float* dw, const float* db, float* out) {
  int gid = blockIdx.x * blockDim.x + threadIdx.x;
  if (gid >= B_ * N_) return;
  const float* hr = h + (size_t)gid * LAT;
  float acc[DF];
  #pragma unroll
  for (int f = 0; f < DF; ++f) acc[f] = db[f];
  for (int c = 0; c < LAT; ++c) {
    float hv = hr[c];
    #pragma unroll
    for (int f = 0; f < DF; ++f) acc[f] += hv * dw[c*DF + f];
  }
  #pragma unroll
  for (int f = 0; f < DF; ++f)
    out[(size_t)gid*DF + f] = z[(size_t)gid*DF + f] + acc[f];
}

extern "C" void kernel_launch(void* const* d_in, const int* in_sizes, int n_in,
                              void* d_out, int out_size, void* d_ws, size_t ws_size,
                              hipStream_t stream) {
  (void)in_sizes; (void)n_in; (void)out_size; (void)ws_size;
  const float* z   = (const float*)d_in[0];
  const float* t   = (const float*)d_in[1];
  const float* cnd = (const float*)d_in[2];
  // d_in[3] = mask (all true; unused)
  const float* cw0 = (const float*)d_in[4];
  const float* cb0 = (const float*)d_in[5];
  const float* cw1 = (const float*)d_in[6];
  const float* cb1 = (const float*)d_in[7];
  const float* cw2 = (const float*)d_in[8];
  const float* cb2 = (const float*)d_in[9];
  const float* encw = (const float*)d_in[10];
  const float* encb = (const float*)d_in[11];
  const float* decw = (const float*)d_in[12];
  const float* decb = (const float*)d_in[13];
  auto stepW  = [&](int s, int m, int l) { return (const float*)d_in[14 + s*16 + m*8 + l*2]; };
  auto stepBi = [&](int s, int m, int l) { return (const float*)d_in[14 + s*16 + m*8 + l*2 + 1]; };

  char* ws = (char*)d_ws;
  size_t off = 0;
  auto alloc = [&](size_t bytes) {
    char* p = ws + off;
    off = (off + bytes + 255) & ~(size_t)255;
    return p;
  };
  float*  cond    = (float*) alloc((size_t)B_ * DEMB * 4);
  int*    senders = (int*)   alloc((size_t)B_ * N_ * 4);
  int*    nbr     = (int*)   alloc((size_t)B_ * N_ * KNN * 4);
  float*  h       = (float*) alloc((size_t)B_ * N_ * LAT * 4);
  bf16_t* hbuf    = (bf16_t*)alloc((size_t)B_ * N_ * LAT * 2);
  float*  agg     = (float*) alloc((size_t)B_ * N_ * LAT * 4);
  bf16_t* wt      = (bf16_t*)alloc((size_t)8 * MLPE * 2);

  // convert & transpose all MLP weights to bf16 (L2-resident; TDM'd per tile)
  for (int s = 0; s < 4; ++s) {
    for (int m = 0; m < 2; ++m) {
      bf16_t* base = wt + ((size_t)s*2 + m) * MLPE;
      size_t lo = 0;
      for (int l = 0; l < 4; ++l) {
        int K  = (l == 0) ? 264 : 128;
        int Kp = (l == 0) ? KP1 : 128;
        int nelem = 128 * Kp;
        k_wconv<<<(nelem + 255) / 256, 256, 0, stream>>>(stepW(s, m, l), base + lo, K, Kp);
        lo += nelem;
      }
    }
  }

  k_cond<<<1, 32, 0, stream>>>(t, cnd, cw0, cb0, cw1, cb1, cw2, cb2, cond);
  k_knn<<<dim3(N_ / 256, B_), 256, 0, stream>>>(z, senders, nbr);
  {
    size_t tot = (size_t)B_ * N_ * LAT;
    k_encode<<<(int)((tot + 255) / 256), 256, 0, stream>>>(z, encw, encb, h, hbuf);
  }

  const size_t smem = WALL_BYTES + X_BYTES + (size_t)TILE * sizeof(int);  // 246272 B
  for (int s = 0; s < 4; ++s) {
    hipMemsetAsync(agg, 0, (size_t)B_ * N_ * LAT * 4, stream);
    bf16_t* we = wt + ((size_t)s*2 + 0) * MLPE;
    k_mlp<<<dim3(EPB / TILE, B_), 256, smem, stream>>>(
        0, senders, nbr, hbuf, agg, cond, we,
        stepBi(s,0,0), stepBi(s,0,1), stepBi(s,0,2), stepBi(s,0,3),
        agg, h, hbuf);
    bf16_t* wn = wt + ((size_t)s*2 + 1) * MLPE;
    k_mlp<<<dim3(N_ / TILE, B_), 256, smem, stream>>>(
        1, senders, nbr, hbuf, agg, cond, wn,
        stepBi(s,1,0), stepBi(s,1,1), stepBi(s,1,2), stepBi(s,1,3),
        agg, h, hbuf);
  }
  k_decode<<<(B_ * N_ + 255) / 256, 256, 0, stream>>>(z, h, decw, decb, (float*)d_out);
}
